// SpikingNet_68289980007056
// MI455X (gfx1250) — matrix operand order repaired and checked by reading, not compiled, not run on hardware
//
#include <hip/hip_runtime.h>

// ---------------------------------------------------------------------------
// SNN forward on MI455X (gfx1250, wave32, WMMA, double-buffered async LDS).
//
//  K0: prep            — fp32 weights -> bf16 in ws, zero recurrent state
//  K1: snn_gemm_cur1   — CUR1(all t) = X @ W1^T  (65536x1024x1024 bf16 WMMA,
//                        4096 blocks; fp32 result stashed in spk1_rec region)
//  per step t (sequential critical path):
//  K2: snn_layer1_step — acc = spk_prev @ Wrec^T; + CUR1[t]; RSynaptic update;
//                        spikes overwrite the CUR1 slice (= final spk1_rec)
//  K3: snn_layer2_step — cur2 = spk1 @ W2^T; Leaky update; spk2/mem2 out
//
// All bf16 tiles are staged global->LDS with GLOBAL_LOAD_ASYNC_TO_LDS_B128
// into double-buffered LDS; s_wait_asynccnt <= S waits only for the older
// batch so the new tile's DMA overlaps the WMMA block.
// ---------------------------------------------------------------------------

#define TT 256
#define BBATCH 256
#define IIN 1024
#define HID 1024
#define OOUT 256
#define LDA 40  // padded LDS row (bf16 elems): 80B stride -> conflict-free frags

typedef __attribute__((ext_vector_type(16))) __bf16 v16bf;
typedef __attribute__((ext_vector_type(8)))  __bf16 v8bf;
typedef __attribute__((ext_vector_type(4)))  __bf16 v4bf;
typedef __attribute__((ext_vector_type(8)))  float  v8f;

union Frag { v16bf v; v8bf h[2]; };

__device__ __forceinline__ v8f wmma_bf16(v16bf a, v16bf b, v8f c) {
  return __builtin_amdgcn_wmma_f32_16x16x32_bf16(false, a, false, b, (short)0, c,
                                                 false, false);
}

// CDNA5 async copy: 16B global -> LDS, tracked by ASYNCcnt (no VGPR data path).
__device__ __forceinline__ void async_ld16(void* lds, const void* gaddr) {
  unsigned l = (unsigned)(unsigned long long)lds;  // low 32 bits = LDS offset
  asm volatile("global_load_async_to_lds_b128 %0, %1, off"
               :: "v"(l), "v"(gaddr) : "memory");
}
template <int N>
__device__ __forceinline__ void async_wait() {
  asm volatile("s_wait_asynccnt %0" :: "i"(N) : "memory");
}

// A/B fragment gather from padded LDS rows (two 16B chunks per lane).
__device__ __forceinline__ Frag a_frag(const __bf16* row, int g) {
  Frag f; const v8bf* p = (const v8bf*)row;
  f.h[0] = p[2 * g];      // K elems {16g..16g+7}
  f.h[1] = p[2 * g + 1];  // K elems {16g+8..16g+15}
  return f;
}
__device__ __forceinline__ Frag b_frag(const __bf16* row, int g) {
  Frag f; const v8bf* p = (const v8bf*)row;
  f.h[0] = p[2 * g];      // K elems {16g..16g+7}
  f.h[1] = p[2 * g + 1];  // K elems {16g+8..16g+15}
  return f;
}

// ---------------------------------------------------------------------------
__global__ __launch_bounds__(256)
void snn_prep(const float* __restrict__ W1, const float* __restrict__ Wrec,
              const float* __restrict__ W2,
              __bf16* __restrict__ W1b, __bf16* __restrict__ Wrb,
              __bf16* __restrict__ W2b,
              float* __restrict__ syn1, float* __restrict__ mem1,
              float* __restrict__ mem2, __bf16* __restrict__ spkA) {
  int i = blockIdx.x * blockDim.x + threadIdx.x;
  if (i < HID * IIN) { W1b[i] = (__bf16)W1[i]; Wrb[i] = (__bf16)Wrec[i]; }
  if (i < OOUT * HID) W2b[i] = (__bf16)W2[i];
  if (i < BBATCH * HID) { syn1[i] = 0.f; mem1[i] = 0.f; spkA[i] = (__bf16)0.f; }
  if (i < BBATCH * OOUT) mem2[i] = 0.f;
}

// ---------------------------------------------------------------------------
// K1: CUR1 = X @ W1^T, M=65536, N=1024, K=1024.
// 256 threads = 8 waves (4 M x 2 N), wave tile 32x64, block tile 128x128.
// B tiles: async double-buffered.  A tiles: global loads issued before the
// WMMA block, cvt+ds_store after it (latency hidden under compute).
// ---------------------------------------------------------------------------
__global__ __launch_bounds__(256)
void snn_gemm_cur1(const float* __restrict__ x,      // (T*B, I) fp32
                   const __bf16* __restrict__ W1b,   // (H, I) bf16
                   float* __restrict__ cur1) {       // (T*B, H) fp32 (d_out)
  __shared__ __align__(16) __bf16 As[2][128 * LDA];
  __shared__ __align__(16) __bf16 Bs[2][128 * LDA];

  const int tid = threadIdx.x, lane = tid & 31, wv = tid >> 5;
  const int g = lane >> 4, ln = lane & 15;
  const int wm = wv & 3, wn = wv >> 2;
  const size_t m0 = (size_t)blockIdx.y * 128;
  const int n0 = blockIdx.x * 128;
  const int r = tid >> 1, lh = tid & 1;

  v8f zero = {0.f, 0.f, 0.f, 0.f, 0.f, 0.f, 0.f, 0.f};
  v8f acc[2][4];
#pragma unroll
  for (int f = 0; f < 2; ++f)
#pragma unroll
    for (int n = 0; n < 4; ++n) acc[f][n] = zero;

  const float* rowA = x + (m0 + r) * IIN + 16 * lh;
  const __bf16* rowB = W1b + (size_t)(n0 + r) * IIN + 16 * lh;
  const unsigned ldsA = r * LDA + 16 * lh;

  // Prologue: stage tile 0 into buffer 0.
  {
    const float4* srcA = (const float4*)rowA;
#pragma unroll
    for (int j = 0; j < 4; ++j) {
      float4 f = srcA[j];
      v4bf p = {(__bf16)f.x, (__bf16)f.y, (__bf16)f.z, (__bf16)f.w};
      *(v4bf*)&As[0][ldsA + 4 * j] = p;
    }
    async_ld16(&Bs[0][ldsA], rowB);
    async_ld16(&Bs[0][ldsA + 8], rowB + 8);
  }

  const int NT = IIN / 32;
  for (int kt = 0; kt < NT; ++kt) {
    const int cur = kt & 1, nxt = cur ^ 1;
    const bool more = (kt + 1 < NT);
    float4 a4[4];
    if (more) {
      const int k1 = (kt + 1) * 32;
      async_ld16(&Bs[nxt][ldsA], rowB + k1);
      async_ld16(&Bs[nxt][ldsA + 8], rowB + k1 + 8);
      const float4* srcA = (const float4*)(rowA + k1);
#pragma unroll
      for (int j = 0; j < 4; ++j) a4[j] = srcA[j];  // issue; use after WMMA
      async_wait<2>();   // only the just-issued pair may remain outstanding
    } else {
      async_wait<0>();
    }
    __syncthreads();

    Frag a0 = a_frag(&As[cur][(32 * wm + ln) * LDA], g);
    Frag a1 = a_frag(&As[cur][(32 * wm + 16 + ln) * LDA], g);
#pragma unroll
    for (int nc = 0; nc < 4; ++nc) {
      Frag b = b_frag(&Bs[cur][(64 * wn + 16 * nc + ln) * LDA], g);
      acc[0][nc] = wmma_bf16(a0.v, b.v, acc[0][nc]);
      acc[1][nc] = wmma_bf16(a1.v, b.v, acc[1][nc]);
    }

    if (more) {  // cvt + store next A tile (loads have had the WMMAs to land)
#pragma unroll
      for (int j = 0; j < 4; ++j) {
        v4bf p = {(__bf16)a4[j].x, (__bf16)a4[j].y, (__bf16)a4[j].z,
                  (__bf16)a4[j].w};
        *(v4bf*)&As[nxt][ldsA + 4 * j] = p;
      }
    }
    __syncthreads();  // all reads of `cur` + stores/DMA into `nxt` ordered
  }

#pragma unroll
  for (int f = 0; f < 2; ++f)
#pragma unroll
    for (int nc = 0; nc < 4; ++nc)
#pragma unroll
      for (int j = 0; j < 8; ++j) {
        const size_t m = m0 + 32 * wm + 16 * f + j + 8 * g;
        const int n = n0 + 64 * wn + 16 * nc + ln;
        cur1[m * HID + n] = acc[f][nc][j];
      }
}

// ---------------------------------------------------------------------------
// K2: recurrent GEMM + RSynaptic update. 128 threads = 4 waves stacked in M,
// wave tile 32x64, block tile 128(B) x 64(H), K = HID.
// Fully async double-buffered staging (6 DMA ops / thread / tile).
// ---------------------------------------------------------------------------
__global__ __launch_bounds__(128)
void snn_layer1_step(const __bf16* __restrict__ Wrb,       // (H,H) bf16
                     const __bf16* __restrict__ spk_prev,  // (B,H) bf16
                     __bf16* __restrict__ spk_new,         // (B,H) bf16
                     float* __restrict__ syn1,             // (B,H) fp32
                     float* __restrict__ mem1,             // (B,H) fp32
                     float* __restrict__ cur_spk,          // d_out slice t
                     const float* __restrict__ alpha_p,
                     const float* __restrict__ th1_p) {
  __shared__ __align__(16) __bf16 As[2][128 * LDA];
  __shared__ __align__(16) __bf16 Bs[2][64 * LDA];

  const int tid = threadIdx.x, lane = tid & 31, wv = tid >> 5;
  const int g = lane >> 4, ln = lane & 15;
  const int b0 = blockIdx.y * 128;
  const int h0 = blockIdx.x * 64;
  const int r = tid >> 1, lh = tid & 1;

  v8f zero = {0.f, 0.f, 0.f, 0.f, 0.f, 0.f, 0.f, 0.f};
  v8f acc[2][4];
#pragma unroll
  for (int f = 0; f < 2; ++f)
#pragma unroll
    for (int n = 0; n < 4; ++n) acc[f][n] = zero;

  const __bf16* rowA = spk_prev + (size_t)(b0 + tid) * HID;
  const __bf16* rowB = Wrb + (size_t)(h0 + r) * HID + 16 * lh;
  const unsigned ldsA = tid * LDA;
  const unsigned ldsB = r * LDA + 16 * lh;

#pragma unroll
  for (int q = 0; q < 4; ++q) async_ld16(&As[0][ldsA + 8 * q], rowA + 8 * q);
  async_ld16(&Bs[0][ldsB], rowB);
  async_ld16(&Bs[0][ldsB + 8], rowB + 8);

  const int NT = HID / 32;
  for (int kt = 0; kt < NT; ++kt) {
    const int cur = kt & 1, nxt = cur ^ 1;
    if (kt + 1 < NT) {
      const int k1 = (kt + 1) * 32;
#pragma unroll
      for (int q = 0; q < 4; ++q)
        async_ld16(&As[nxt][ldsA + 8 * q], rowA + k1 + 8 * q);
      async_ld16(&Bs[nxt][ldsB], rowB + k1);
      async_ld16(&Bs[nxt][ldsB + 8], rowB + k1 + 8);
      async_wait<6>();
    } else {
      async_wait<0>();
    }
    __syncthreads();

    Frag a0 = a_frag(&As[cur][(32 * wv + ln) * LDA], g);
    Frag a1 = a_frag(&As[cur][(32 * wv + 16 + ln) * LDA], g);
#pragma unroll
    for (int nc = 0; nc < 4; ++nc) {
      Frag b = b_frag(&Bs[cur][(16 * nc + ln) * LDA], g);
      acc[0][nc] = wmma_bf16(a0.v, b.v, acc[0][nc]);
      acc[1][nc] = wmma_bf16(a1.v, b.v, acc[1][nc]);
    }
    __syncthreads();
  }

  const float alpha = *alpha_p;
  const float th1 = *th1_p;
#pragma unroll
  for (int f = 0; f < 2; ++f)
#pragma unroll
    for (int nc = 0; nc < 4; ++nc)
#pragma unroll
      for (int j = 0; j < 8; ++j) {
        const int b = b0 + 32 * wv + 16 * f + j + 8 * g;
        const int h = h0 + 16 * nc + ln;
        const size_t idx = (size_t)b * HID + h;
        const float cur = cur_spk[idx] + acc[f][nc][j];  // CUR1 + recurrent
        const float s = alpha * syn1[idx] + cur;
        const float mo = mem1[idx];
        const float reset = (mo - th1) > 0.f ? 1.f : 0.f;
        float mn = 0.9f * mo + s - reset * th1;
        const float spk = (mn - th1) > 0.f ? 1.f : 0.f;
        mn -= (spk - reset) * th1;
        syn1[idx] = s;
        mem1[idx] = mn;
        spk_new[idx] = (__bf16)spk;
        cur_spk[idx] = spk;  // overwrite CUR1 with recorded spike
      }
}

// ---------------------------------------------------------------------------
// K3: cur2 = spk1 @ W2^T + Leaky update. Block tile 128(B) x 64(O), K = HID.
// ---------------------------------------------------------------------------
__global__ __launch_bounds__(128)
void snn_layer2_step(const __bf16* __restrict__ spk1,  // (B,H) bf16
                     const __bf16* __restrict__ W2b,   // (O,H) bf16
                     float* __restrict__ mem2,         // (B,O) fp32
                     float* __restrict__ spk2_rec,     // d_out slice t
                     float* __restrict__ mem2_rec) {   // d_out slice t
  __shared__ __align__(16) __bf16 As[2][128 * LDA];
  __shared__ __align__(16) __bf16 Bs[2][64 * LDA];

  const int tid = threadIdx.x, lane = tid & 31, wv = tid >> 5;
  const int g = lane >> 4, ln = lane & 15;
  const int b0 = blockIdx.y * 128;
  const int o0 = blockIdx.x * 64;
  const int r = tid >> 1, lh = tid & 1;

  v8f zero = {0.f, 0.f, 0.f, 0.f, 0.f, 0.f, 0.f, 0.f};
  v8f acc[2][4];
#pragma unroll
  for (int f = 0; f < 2; ++f)
#pragma unroll
    for (int n = 0; n < 4; ++n) acc[f][n] = zero;

  const __bf16* rowA = spk1 + (size_t)(b0 + tid) * HID;
  const __bf16* rowB = W2b + (size_t)(o0 + r) * HID + 16 * lh;
  const unsigned ldsA = tid * LDA;
  const unsigned ldsB = r * LDA + 16 * lh;

#pragma unroll
  for (int q = 0; q < 4; ++q) async_ld16(&As[0][ldsA + 8 * q], rowA + 8 * q);
  async_ld16(&Bs[0][ldsB], rowB);
  async_ld16(&Bs[0][ldsB + 8], rowB + 8);

  const int NT = HID / 32;
  for (int kt = 0; kt < NT; ++kt) {
    const int cur = kt & 1, nxt = cur ^ 1;
    if (kt + 1 < NT) {
      const int k1 = (kt + 1) * 32;
#pragma unroll
      for (int q = 0; q < 4; ++q)
        async_ld16(&As[nxt][ldsA + 8 * q], rowA + k1 + 8 * q);
      async_ld16(&Bs[nxt][ldsB], rowB + k1);
      async_ld16(&Bs[nxt][ldsB + 8], rowB + k1 + 8);
      async_wait<6>();
    } else {
      async_wait<0>();
    }
    __syncthreads();

    Frag a0 = a_frag(&As[cur][(32 * wv + ln) * LDA], g);
    Frag a1 = a_frag(&As[cur][(32 * wv + 16 + ln) * LDA], g);
#pragma unroll
    for (int nc = 0; nc < 4; ++nc) {
      Frag b = b_frag(&Bs[cur][(16 * nc + ln) * LDA], g);
      acc[0][nc] = wmma_bf16(a0.v, b.v, acc[0][nc]);
      acc[1][nc] = wmma_bf16(a1.v, b.v, acc[1][nc]);
    }
    __syncthreads();
  }

#pragma unroll
  for (int f = 0; f < 2; ++f)
#pragma unroll
    for (int nc = 0; nc < 4; ++nc)
#pragma unroll
      for (int j = 0; j < 8; ++j) {
        const int b = b0 + 32 * wv + 16 * f + j + 8 * g;
        const int o = o0 + 16 * nc + ln;
        const size_t idx = (size_t)b * OOUT + o;
        const float mo = mem2[idx];
        const float reset = (mo - 1.f) > 0.f ? 1.f : 0.f;
        const float mn = 0.9f * mo + acc[f][nc][j] - reset;
        const float spk = (mn - 1.f) > 0.f ? 1.f : 0.f;
        mem2[idx] = mn;
        spk2_rec[idx] = spk;
        mem2_rec[idx] = mn;
      }
}

// ---------------------------------------------------------------------------
extern "C" void kernel_launch(void* const* d_in, const int* in_sizes, int n_in,
                              void* d_out, int out_size, void* d_ws,
                              size_t ws_size, hipStream_t stream) {
  (void)in_sizes; (void)n_in; (void)out_size; (void)ws_size;

  const float* x     = (const float*)d_in[0];
  const float* W1    = (const float*)d_in[1];
  const float* Wrec  = (const float*)d_in[2];
  const float* W2    = (const float*)d_in[3];
  const float* alpha = (const float*)d_in[4];
  const float* th1   = (const float*)d_in[5];

  char* ws = (char*)d_ws;
  __bf16* W1b  = (__bf16*)(ws + 0);           // 2 MB
  __bf16* Wrb  = (__bf16*)(ws + (2u << 20));  // 2 MB
  __bf16* W2b  = (__bf16*)(ws + (4u << 20));  // 512 KB
  float*  syn1 = (float*)(ws + 4718592);      // 1 MB
  float*  mem1 = (float*)(ws + 5767168);      // 1 MB
  float*  mem2 = (float*)(ws + 6815744);      // 256 KB
  __bf16* spkA = (__bf16*)(ws + 7077888);     // 512 KB
  __bf16* spkB = (__bf16*)(ws + 7602176);     // 512 KB

  float* out      = (float*)d_out;
  float* out_spk2 = out;                                   // (T,B,O)
  float* out_mem2 = out + (size_t)TT * BBATCH * OOUT;      // (T,B,O)
  float* out_spk1 = out + 2 * (size_t)TT * BBATCH * OOUT;  // (T,B,H) (CUR1 scratch)

  snn_prep<<<(HID * IIN + 255) / 256, 256, 0, stream>>>(
      W1, Wrec, W2, W1b, Wrb, W2b, syn1, mem1, mem2, spkA);

  // Bulk input projection for all timesteps (fills spk1_rec region with CUR1).
  snn_gemm_cur1<<<dim3(HID / 128, (TT * BBATCH) / 128), 256, 0, stream>>>(
      x, W1b, out_spk1);

  const dim3 blk(128);
  const dim3 grid1(HID / 64, BBATCH / 128);   // (16, 2)
  const dim3 grid2(OOUT / 64, BBATCH / 128);  // (4, 2)

  for (int t = 0; t < TT; ++t) {
    const __bf16* sp = (t & 1) ? spkB : spkA;
    __bf16*       sn = (t & 1) ? spkA : spkB;
    snn_layer1_step<<<grid1, blk, 0, stream>>>(
        Wrb, sp, sn, syn1, mem1, out_spk1 + (size_t)t * BBATCH * HID, alpha,
        th1);
    snn_layer2_step<<<grid2, blk, 0, stream>>>(
        sn, W2b, mem2, out_spk2 + (size_t)t * BBATCH * OOUT,
        out_mem2 + (size_t)t * BBATCH * OOUT);
  }
}
